// a_decoder_35811437314185
// MI455X (gfx1250) — matrix operand order repaired and checked by reading, not compiled, not run on hardware
//
#include <hip/hip_runtime.h>
#include <math.h>

#define NEGC (-1000000000.0f)
#define B_  128
#define N_  512
#define D_  256

typedef __attribute__((ext_vector_type(16))) __bf16 v16bf;
typedef __attribute__((ext_vector_type(8)))  float  v8f;

// ---------------------------------------------------------------- bf16 split
__device__ __forceinline__ void split_bf16(float x, __bf16& hi, __bf16& lo) {
  hi = (__bf16)x;
  lo = (__bf16)(x - (float)hi);
}

// A fragment: 16x32 bf16 (ISA 7.12.2 layout). lane m = lane&15, half = lane>>4,
// VGPR v holds K = (v<4?0:16) + half*8 + (v&3)*2 , +1
__device__ __forceinline__ void load_a_frag(const float* sA, int lda, int r0,
                                            v16bf& ahi, v16bf& alo) {
  int lane = threadIdx.x & 31;
  int m = lane & 15;
  int half = lane >> 4;
  const float* rowp = sA + (r0 + m) * lda;
#pragma unroll
  for (int v = 0; v < 8; ++v) {
    int kb = ((v < 4) ? 0 : 16) + half * 8 + (v & 3) * 2;
    __bf16 h0, l0, h1, l1;
    split_bf16(rowp[kb], h0, l0);
    split_bf16(rowp[kb + 1], h1, l1);
    ahi[2 * v] = h0; ahi[2 * v + 1] = h1;
    alo[2 * v] = l0; alo[2 * v + 1] = l1;
  }
}

// B fragment: 32x16 bf16. lane n = lane&15, half = lane>>4,
// VGPR v holds K = (v<4?0:16) + half*8 + (v&3)*2 , +1
__device__ __forceinline__ void load_b_frag(const float* sB, int ldb, int c0,
                                            v16bf& bhi, v16bf& blo) {
  int lane = threadIdx.x & 31;
  int n = lane & 15;
  int half = lane >> 4;
#pragma unroll
  for (int v = 0; v < 8; ++v) {
    int k = ((v < 4) ? 0 : 16) + half * 8 + (v & 3) * 2;
    __bf16 h0, l0, h1, l1;
    split_bf16(sB[k * ldb + c0 + n], h0, l0);
    split_bf16(sB[(k + 1) * ldb + c0 + n], h1, l1);
    bhi[2 * v] = h0; bhi[2 * v + 1] = h1;
    blo[2 * v] = l0; blo[2 * v + 1] = l1;
  }
}

// split-precision product: C += Ah*Bh + Ah*Bl + Al*Bh  (~fp32 fidelity)
__device__ __forceinline__ v8f mma3(v16bf ah, v16bf al, v16bf bh, v16bf bl, v8f c) {
  c = __builtin_amdgcn_wmma_f32_16x16x32_bf16(false, ah, false, bh, (short)0, c, false, false);
  c = __builtin_amdgcn_wmma_f32_16x16x32_bf16(false, ah, false, bl, (short)0, c, false, false);
  c = __builtin_amdgcn_wmma_f32_16x16x32_bf16(false, al, false, bh, (short)0, c, false, false);
  return c;
}

// ---------------------------------------------------------------- init
__global__ __launch_bounds__(256) void dec_init_kernel(const float* __restrict__ av,
                                                       float* __restrict__ query,
                                                       float* __restrict__ ring,
                                                       unsigned char* __restrict__ mask,
                                                       unsigned* __restrict__ bar) {
  int i = blockIdx.x * 256 + threadIdx.x;       // 65536 threads
  if (i < B_ * N_) mask[i] = 0;
  if (i < B_ * D_) {
    int b = i >> 8, d = i & 255;
    query[i] = av[((size_t)b * N_ + 0) * D_ + d];
  }
  if (i < B_ * 3 * D_) ring[i] = 0.0f;
  if (i < 2) bar[i] = 0u;
}

// ---------------------------------------------------------------- act_t GEMM
// C[M,256] = A[M,256] @ W[256,256] + bias ; 64x64 tile per block, 8 waves
__global__ __launch_bounds__(256) void act_gemm_kernel(const float* __restrict__ A,
                                                       const float* __restrict__ W,
                                                       const float* __restrict__ bias,
                                                       float* __restrict__ C) {
  __shared__ float sA[64][33];
  __shared__ float sB[32][68];
  int bx = blockIdx.x;
  int colb = (bx & 3) * 64;
  int rowb = (bx >> 2) * 64;
  int tid = threadIdx.x;
  int wave = tid >> 5;
  int rt = wave & 3;
  int ct0 = (wave >> 2) * 2;
  v8f acc0 = {}; v8f acc1 = {};
  for (int k0 = 0; k0 < 256; k0 += 32) {
    __syncthreads();
#pragma unroll
    for (int e = 0; e < 8; ++e) {
      int i = tid + e * 256;
      int r = i >> 5, c = i & 31;
      sA[r][c] = A[(size_t)(rowb + r) * 256 + k0 + c];
    }
#pragma unroll
    for (int e = 0; e < 8; ++e) {
      int i = tid + e * 256;
      int r = i >> 6, c = i & 63;
      sB[r][c] = W[(size_t)(k0 + r) * 256 + colb + c];
    }
    __syncthreads();
    v16bf ah, al, bh, bl;
    load_a_frag(&sA[0][0], 33, rt * 16, ah, al);
    load_b_frag(&sB[0][0], 68, ct0 * 16, bh, bl);
    acc0 = mma3(ah, al, bh, bl, acc0);
    load_b_frag(&sB[0][0], 68, (ct0 + 1) * 16, bh, bl);
    acc1 = mma3(ah, al, bh, bl, acc1);
  }
  int lane = tid & 31;
  int n = lane & 15;
  int mb = (lane >> 4) * 8;
#pragma unroll
  for (int v = 0; v < 8; ++v) {
    int row = rowb + rt * 16 + mb + v;
    int c0 = colb + ct0 * 16 + n;
    C[(size_t)row * 256 + c0]      = acc0[v] + bias[c0];
    C[(size_t)row * 256 + c0 + 16] = acc1[v] + bias[c0 + 16];
  }
}

// ---------------------------------------------------------------- grid barrier
__device__ __forceinline__ void grid_barrier(unsigned* bar, unsigned nblk) {
  __syncthreads();
  if (threadIdx.x == 0) {
    __threadfence();
    volatile unsigned* vgen = bar + 1;
    unsigned g = *vgen;
    if (atomicAdd(bar, 1u) == nblk - 1u) {
      atomicExch(bar, 0u);
      __threadfence();
      atomicAdd((unsigned*)(bar + 1), 1u);
    } else {
      while (*vgen == g) { __builtin_amdgcn_s_sleep(2); }
    }
    __threadfence();
  }
  __syncthreads();
}

// ---------------------------------------------------------------- persistent decoder
__global__ __launch_bounds__(256) void decode_persistent_kernel(
    const float* __restrict__ av,     // [B,N,D]
    const float* __restrict__ actt,   // [B,N,D]
    const float* __restrict__ Wq, const float* __restrict__ bq,
    const float* __restrict__ vw, const float* __restrict__ vb,
    const float* __restrict__ W1, const float* __restrict__ b1,
    const float* __restrict__ W2, const float* __restrict__ b2,
    const float* __restrict__ W3, const float* __restrict__ b3,
    float* __restrict__ query, float* __restrict__ ring,
    unsigned char* __restrict__ mask,
    float* __restrict__ scoresL, float* __restrict__ rowmax, float* __restrict__ rowsum,
    unsigned* __restrict__ bar,
    float* __restrict__ out_map, float* __restrict__ out_prob) {
  __shared__ float sq[256];
  __shared__ float sqt[256];
  __shared__ float sscores[512];
  __shared__ float sval[256];
  __shared__ int   sidx[256];
  __shared__ float gA[16][33];
  __shared__ float gB[32][17];
  __shared__ int   sIdxSel;
  __shared__ int   sidxf[128];

  const int b = blockIdx.x;     // one block per batch row (also = GEMM tile id)
  const int tid = threadIdx.x;
  const int wave = tid >> 5;
  const int lane = tid & 31;
  const float vb0 = vb[0];

  float vwreg[8];
#pragma unroll
  for (int j = 0; j < 8; ++j) vwreg[j] = vw[lane + 32 * j];

  for (int t = 0; t < N_; ++t) {
    // ----- Phase A: q_t row, tanh scores, argmax, selection -----
    sq[tid] = query[(size_t)b * D_ + tid];
    __syncthreads();
    float qa = bq[tid];
#pragma unroll 4
    for (int k = 0; k < 256; ++k) qa = fmaf(sq[k], Wq[(size_t)k * 256 + tid], qa);
    sqt[tid] = qa;
    __syncthreads();

    float qreg[8];
#pragma unroll
    for (int j = 0; j < 8; ++j) qreg[j] = sqt[lane + 32 * j];

    for (int n = wave; n < N_; n += 8) {
      const float* ap = actt + ((size_t)b * N_ + n) * D_;
      float p = 0.0f;
#pragma unroll
      for (int j = 0; j < 8; ++j) {
        int d = lane + 32 * j;
        p += vwreg[j] * tanhf(ap[d] + qreg[j]);
      }
#pragma unroll
      for (int off = 16; off; off >>= 1) p += __shfl_xor(p, off, 32);
      if (lane == 0) {
        float s = p + vb0;
        if (mask[(size_t)b * N_ + n]) s = NEGC;
        sscores[n] = s;
      }
    }
    __syncthreads();

    // argmax (lowest index on tie, matching jnp.argmax)
    {
      float v0 = sscores[tid], v1 = sscores[tid + 256];
      float bv = v0; int bi = tid;
      if (v1 > v0) { bv = v1; bi = tid + 256; }
      sval[tid] = bv; sidx[tid] = bi;
      __syncthreads();
      for (int s = 128; s; s >>= 1) {
        if (tid < s) {
          float ov = sval[tid + s]; int oi = sidx[tid + s];
          if (ov > sval[tid] || (ov == sval[tid] && oi < sidx[tid])) {
            sval[tid] = ov; sidx[tid] = oi;
          }
        }
        __syncthreads();
      }
      if (tid == 0) sIdxSel = sidx[0];
      __syncthreads();
    }
    const int sel = sIdxSel;
    const float smax = sval[0];

    if (tid == 0) {
      mask[(size_t)b * N_ + sel] = 1;
      out_map[(size_t)b * N_ + t] = (float)sel;  // mapping_seq[b][t]
    }
    // gather chosen action vector into ring slot t%3 (newest)
    ring[(size_t)b * (3 * D_) + (t % 3) * D_ + tid] =
        av[((size_t)b * N_ + sel) * D_ + tid];

    if (t == N_ - 1) {  // store softmax stats for final prob matrix
      float es = expf(sscores[tid] - smax) + expf(sscores[tid + 256] - smax);
      __syncthreads();
      sval[tid] = es;
      __syncthreads();
      for (int s = 128; s; s >>= 1) {
        if (tid < s) sval[tid] += sval[tid + s];
        __syncthreads();
      }
      if (tid == 0) { rowmax[b] = smax; rowsum[b] = sval[0]; }
      scoresL[(size_t)b * N_ + tid]       = sscores[tid];
      scoresL[(size_t)b * N_ + tid + 256] = sscores[tid + 256];
    }

    grid_barrier(bar, B_);

    // ----- Phase B: query = relu(v_t@W1 + v_{t-1}@W2 + v_{t-2}@W3 + b123) -----
    if (t < N_ - 1) {
      const int rt = b >> 4;   // 8 row tiles of 16
      const int ct = b & 15;   // 16 col tiles of 16
      const int slots[3] = { t % 3, (t + 2) % 3, (t + 1) % 3 };
      const float* Ws[3] = { W1, W2, W3 };
      v8f acc = {};
      for (int kc = 0; kc < 3; ++kc) {
        const float* Wc = Ws[kc];
        const int slot = slots[kc];
        for (int kk = 0; kk < 256; kk += 32) {
          __syncthreads();
#pragma unroll
          for (int e = 0; e < 2; ++e) {
            int i = tid + e * 256;
            int r = i >> 5, c = i & 31;
            gA[r][c] = ring[(size_t)(rt * 16 + r) * (3 * D_) + slot * D_ + kk + c];
          }
#pragma unroll
          for (int e = 0; e < 2; ++e) {
            int i = tid + e * 256;
            int r = i >> 4, c = i & 15;
            gB[r][c] = Wc[(size_t)(kk + r) * 256 + ct * 16 + c];
          }
          __syncthreads();
          if (tid < 32) {   // wave 0 does WMMA with full EXEC
            v16bf ah, al, bh, bl;
            load_a_frag(&gA[0][0], 33, 0, ah, al);
            load_b_frag(&gB[0][0], 17, 0, bh, bl);
            acc = mma3(ah, al, bh, bl, acc);
          }
        }
      }
      if (tid < 32) {
        int n = lane & 15, mb = (lane >> 4) * 8;
#pragma unroll
        for (int v = 0; v < 8; ++v) {
          int row = rt * 16 + mb + v;
          int col = ct * 16 + n;
          float val = acc[v] + b1[col] + b2[col] + b3[col];
          query[(size_t)row * D_ + col] = fmaxf(val, 0.0f);
        }
      }
      grid_barrier(bar, B_);
    }
  }

  // ----- final: prob[i][j] = probs[i, idx_final[j]] -----
  if (tid < 128) sidxf[tid] = (int)out_map[(size_t)tid * N_ + (N_ - 1)];
  __syncthreads();
  if (tid < 128) {
    float m = rowmax[b], s = rowsum[b];
    int ij = sidxf[tid];
    out_prob[(size_t)b * B_ + tid] = expf(scoresL[(size_t)b * N_ + ij] - m) / s;
  }
}

// ---------------------------------------------------------------- launch
extern "C" void kernel_launch(void* const* d_in, const int* in_sizes, int n_in,
                              void* d_out, int out_size, void* d_ws, size_t ws_size,
                              hipStream_t stream) {
  (void)in_sizes; (void)n_in; (void)out_size; (void)ws_size;
  const float* av   = (const float*)d_in[0];
  const float* Wref = (const float*)d_in[1];
  const float* bref = (const float*)d_in[2];
  const float* Wq   = (const float*)d_in[3];
  const float* bq   = (const float*)d_in[4];
  const float* vw   = (const float*)d_in[5];
  const float* vb   = (const float*)d_in[6];
  const float* W1   = (const float*)d_in[7];
  const float* b1   = (const float*)d_in[8];
  const float* W2   = (const float*)d_in[9];
  const float* b2   = (const float*)d_in[10];
  const float* W3   = (const float*)d_in[11];
  const float* b3   = (const float*)d_in[12];

  float* out_map  = (float*)d_out;                 // B*N
  float* out_prob = (float*)d_out + B_ * N_;       // B*B

  // workspace carve-up (256B aligned)
  char* w = (char*)d_ws;
  size_t off = 0;
  auto take = [&](size_t bytes) {
    void* p = w + off;
    off = (off + bytes + 255) & ~(size_t)255;
    return p;
  };
  float* actt   = (float*)take((size_t)B_ * N_ * D_ * sizeof(float)); // 64 MB (L2-resident)
  float* query  = (float*)take((size_t)B_ * D_ * sizeof(float));
  float* ring   = (float*)take((size_t)B_ * 3 * D_ * sizeof(float));
  float* scoresL= (float*)take((size_t)B_ * N_ * sizeof(float));
  float* rowmax = (float*)take((size_t)B_ * sizeof(float));
  float* rowsum = (float*)take((size_t)B_ * sizeof(float));
  unsigned char* mask = (unsigned char*)take((size_t)B_ * N_);
  unsigned* bar = (unsigned*)take(2 * sizeof(unsigned));

  dec_init_kernel<<<256, 256, 0, stream>>>(av, query, ring, mask, bar);

  // act_t = AV @ Wref + bref : M=65536, 64x64 tiles -> 1024*4 blocks
  act_gemm_kernel<<<(B_ * N_ / 64) * (D_ / 64), 256, 0, stream>>>(av, Wref, bref, actt);

  decode_persistent_kernel<<<B_, 256, 0, stream>>>(
      av, actt, Wq, bq, vw, vb, W1, b1, W2, b2, W3, b3,
      query, ring, mask, scoresL, rowmax, rowsum, bar, out_map, out_prob);
}
